// Attention_50345606644062
// MI455X (gfx1250) — matrix operand order repaired and checked by reading, not compiled, not run on hardware
//
#include <hip/hip_runtime.h>
#include <hip/hip_bf16.h>
#include <math.h>

typedef __bf16 bf16;
typedef __attribute__((ext_vector_type(16))) __bf16 v16bf;
typedef __attribute__((ext_vector_type(8)))  __bf16 v8bf;
typedef __attribute__((ext_vector_type(8)))  float   v8f;

#define DIMC   192
#define HEADS  8
#define HD     24
#define BATCH  8
#define IMH    128
#define IMW    128
#define HW     (IMH*IMW)     /* 16384 */
#define C2     (2*DIMC)      /* 384 */

// CDNA5 async global->LDS copy (ASYNCcnt path). The builtin's params are
// pointers to 16-byte int vectors: AS1 for global, AS3 for LDS (clang printed
// the first param as "int __vector(4) __device__ *").
#if defined(__has_builtin)
#if __has_builtin(__builtin_amdgcn_global_load_async_to_lds_b128) && \
    __has_builtin(__builtin_amdgcn_s_wait_asynccnt)
#define HAVE_ASYNC_LDS 1
#endif
#endif

#if defined(HAVE_ASYNC_LDS)
typedef int cp_v4i __attribute__((vector_size(16)));
typedef __attribute__((address_space(1))) cp_v4i g_v4i;
typedef __attribute__((address_space(3))) cp_v4i l_v4i;
#define ASYNC_CP_B128(gp, lp) \
  __builtin_amdgcn_global_load_async_to_lds_b128((g_v4i*)(gp), (l_v4i*)(lp), 0, 0)
#define ASYNC_WAIT() __builtin_amdgcn_s_wait_asynccnt(0)
#endif

// ---------------------------------------------------------------------------
// WMMA fragment loaders (wave32); A and B share one contiguous k order (k = 16h + j).
// A (16x32 bf16, MxK): lane L -> row m=L&15; halves j: k = kb + j,
//   kb = (L<16)?0:16.  Stored in LDS row-major ld=32 -> two 16B contiguous loads.
// B (32x16 bf16, KxN): lane L -> col n=L&15; halves j: k = kb + j, kb=(L<16)?0:16.
//   Stored in LDS *transposed* [n][k] ld=32 -> one 32B contiguous load.
// C/D (16x16 f32): VGPR r -> m = r + ((L<16)?0:8), n = L&15.
// ---------------------------------------------------------------------------
__device__ inline v16bf lds_a_frag(const bf16* At, int lane) {
  const int m  = lane & 15;
  const int kb = (lane < 16) ? 0 : 16;
  const bf16* p = At + m * 32 + kb;
  v8bf lo = *(const v8bf*)(p);
  v8bf hi = *(const v8bf*)(p + 8);
  v16bf r;
#pragma unroll
  for (int i = 0; i < 8; ++i) { r[i] = lo[i]; r[i + 8] = hi[i]; }
  return r;
}

__device__ inline v16bf lds_b_frag(const bf16* Bt, int lane) {
  const int n  = lane & 15;
  const int kb = (lane < 16) ? 0 : 16;
  return *(const v16bf*)(Bt + n * 32 + kb);
}

__device__ inline v8f wmma_bf16(v16bf a, v16bf b, v8f c) {
  return __builtin_amdgcn_wmma_f32_16x16x32_bf16(false, a, false, b,
                                                 (short)0, c, false, false);
}

// pack two f32 -> one 32-bit word of two bf16 (compiler emits v_cvt_pk_bf16_f32)
__device__ inline unsigned pack_bf16x2(float a, float b) {
  union { bf16 h[2]; unsigned u; } pk;
  pk.h[0] = (bf16)a; pk.h[1] = (bf16)b;
  return pk.u;
}

// ---------------------------------------------------------------------------
// f32 -> bf16 weight conversion
// ---------------------------------------------------------------------------
__global__ void cvt_f32_bf16(const float* __restrict__ s, bf16* __restrict__ d, int n) {
  int i = blockIdx.x * 256 + threadIdx.x;
  if (i < n) d[i] = (bf16)s[i];
}

// ---------------------------------------------------------------------------
// 1x1 conv as GEMM: out[b,co,p] = sum_ci W[co,ci] * in[b,ci,p]
// grid.x = B*128 (128-pixel tiles), grid.y = CO/64, block = 256 (8 waves 2x4)
// A tile (bf16 weights, layout-exact) copied with async global->LDS; B tile
// transposed + converted in VGPRs with packed b32 LDS stores.
// ---------------------------------------------------------------------------
__global__ void __launch_bounds__(256)
conv1x1_wmma(const float* __restrict__ in, const bf16* __restrict__ wbf,
             float* __restrict__ out, int CI, int CO) {
  __shared__ __align__(16) bf16 Abuf[64 * 32];
  __shared__ __align__(16) bf16 Bt[128 * 32];
  const int t = threadIdx.x, lane = t & 31, wv = t >> 5;
  const int b   = blockIdx.x >> 7;
  const int p0  = (blockIdx.x & 127) * 128;
  const int co0 = blockIdx.y * 64;
  const float* inb  = in  + (size_t)b * CI * HW;
  float*       outb = out + (size_t)b * CO * HW;

  const v8f zero8 = {0.f, 0.f, 0.f, 0.f, 0.f, 0.f, 0.f, 0.f};
  v8f acc[2][2];
#pragma unroll
  for (int f = 0; f < 2; ++f)
#pragma unroll
    for (int g = 0; g < 2; ++g) acc[f][g] = zero8;

  const int mi = wv >> 2, ni = wv & 3;

  for (int k0 = 0; k0 < CI; k0 += 32) {
    // A tile: 64x32 bf16 weights, 16B per thread
    {
      const int m = t >> 2, kk = (t & 3) * 8;
      const bf16* gp = wbf + (size_t)(co0 + m) * CI + k0 + kk;
      bf16* lp = &Abuf[m * 32 + kk];
#if defined(HAVE_ASYNC_LDS)
      ASYNC_CP_B128(gp, lp);        // overlaps with B staging below
#else
      *(v8bf*)lp = *(const v8bf*)gp;
#endif
    }
    // B tile: 32x128 f32 -> transposed bf16 [n][k], packed 32-bit LDS stores
    if (k0 + 32 < CI) __builtin_prefetch(inb + (size_t)(k0 + 32) * HW + p0, 0, 1);
#pragma unroll
    for (int r = 0; r < 8; ++r) {
      const int linear = t + 256 * r;          // 128 n x 16 k-pairs
      const int kp = linear >> 7, n = linear & 127;
      const float v0 = inb[(size_t)(k0 + 2 * kp    ) * HW + p0 + n];
      const float v1 = inb[(size_t)(k0 + 2 * kp + 1) * HW + p0 + n];
      *(unsigned*)&Bt[n * 32 + 2 * kp] = pack_bf16x2(v0, v1);
    }
#if defined(HAVE_ASYNC_LDS)
    ASYNC_WAIT();
#endif
    __syncthreads();
    v16bf a0 = lds_a_frag(Abuf + (mi * 32     ) * 32, lane);
    v16bf a1 = lds_a_frag(Abuf + (mi * 32 + 16) * 32, lane);
    v16bf b0 = lds_b_frag(Bt   + (ni * 32     ) * 32, lane);
    v16bf b1 = lds_b_frag(Bt   + (ni * 32 + 16) * 32, lane);
    acc[0][0] = wmma_bf16(a0, b0, acc[0][0]);
    acc[0][1] = wmma_bf16(a0, b1, acc[0][1]);
    acc[1][0] = wmma_bf16(a1, b0, acc[1][0]);
    acc[1][1] = wmma_bf16(a1, b1, acc[1][1]);
    __syncthreads();
  }

  const int n  = lane & 15;
  const int mh = (lane < 16) ? 0 : 8;
#pragma unroll
  for (int f = 0; f < 2; ++f)
#pragma unroll
    for (int g = 0; g < 2; ++g)
#pragma unroll
      for (int r = 0; r < 8; ++r)
        outb[(size_t)(co0 + mi * 32 + f * 16 + mh + r) * HW +
             p0 + ni * 32 + g * 16 + n] = acc[f][g][r];
}

// ---------------------------------------------------------------------------
// Dense 3x3 conv as implicit GEMM (K = 192*9), SAME zero padding.
// grid.x = B*H (one image row = one 128-pixel N tile), grid.y = 3 (CO/64)
// ---------------------------------------------------------------------------
__global__ void __launch_bounds__(256)
conv3x3_wmma(const float* __restrict__ in, const bf16* __restrict__ wbf,
             float* __restrict__ out) {
  __shared__ __align__(16) bf16 Abuf[64 * 32];
  __shared__ __align__(16) bf16 Bt[128 * 32];
  const int t = threadIdx.x, lane = t & 31, wv = t >> 5;
  const int b = blockIdx.x >> 7, h = blockIdx.x & 127;
  const int co0 = blockIdx.y * 64;
  const float* inb = in + (size_t)b * DIMC * HW;

  const v8f zero8 = {0.f, 0.f, 0.f, 0.f, 0.f, 0.f, 0.f, 0.f};
  v8f acc[2][2];
#pragma unroll
  for (int f = 0; f < 2; ++f)
#pragma unroll
    for (int g = 0; g < 2; ++g) acc[f][g] = zero8;

  const int mi = wv >> 2, ni = wv & 3;

  for (int tap = 0; tap < 9; ++tap) {
    const int dy = tap / 3 - 1, dx = tap % 3 - 1;
    const int sh = h + dy;
    const bool hok = ((unsigned)sh < (unsigned)IMH);
    for (int k0 = 0; k0 < DIMC; k0 += 32) {
      // A tile: w[co, ci, tap] (OIHW, stride 9 over tap) -> scalar gathers
      {
        const int m = t >> 2, kk = (t & 3) * 8;
#pragma unroll
        for (int i = 0; i < 8; ++i)
          Abuf[m * 32 + kk + i] =
              wbf[((size_t)(co0 + m) * DIMC + (k0 + kk + i)) * 9 + tap];
      }
      // B tile: shifted input row, zero padded, packed b32 LDS stores
#pragma unroll
      for (int r = 0; r < 8; ++r) {
        const int linear = t + 256 * r;
        const int kp = linear >> 7, n = linear & 127;
        const int sw = n + dx;
        float v0 = 0.f, v1 = 0.f;
        if (hok && (unsigned)sw < (unsigned)IMW) {
          const float* base = inb + sh * IMW + sw;
          v0 = base[(size_t)(k0 + 2 * kp    ) * HW];
          v1 = base[(size_t)(k0 + 2 * kp + 1) * HW];
        }
        *(unsigned*)&Bt[n * 32 + 2 * kp] = pack_bf16x2(v0, v1);
      }
      __syncthreads();
      v16bf a0 = lds_a_frag(Abuf + (mi * 32     ) * 32, lane);
      v16bf a1 = lds_a_frag(Abuf + (mi * 32 + 16) * 32, lane);
      v16bf b0 = lds_b_frag(Bt   + (ni * 32     ) * 32, lane);
      v16bf b1 = lds_b_frag(Bt   + (ni * 32 + 16) * 32, lane);
      acc[0][0] = wmma_bf16(a0, b0, acc[0][0]);
      acc[0][1] = wmma_bf16(a0, b1, acc[0][1]);
      acc[1][0] = wmma_bf16(a1, b0, acc[1][0]);
      acc[1][1] = wmma_bf16(a1, b1, acc[1][1]);
      __syncthreads();
    }
  }

  const int p0 = h * IMW;
  const int n  = lane & 15;
  const int mh = (lane < 16) ? 0 : 8;
  float* outb = out + (size_t)b * DIMC * HW;
#pragma unroll
  for (int f = 0; f < 2; ++f)
#pragma unroll
    for (int g = 0; g < 2; ++g)
#pragma unroll
      for (int r = 0; r < 8; ++r)
        outb[(size_t)(co0 + mi * 32 + f * 16 + mh + r) * HW +
             p0 + ni * 32 + g * 16 + n] = acc[f][g][r];
}

// ---------------------------------------------------------------------------
// Depthwise 3x3 (groups = 384), f32, memory bound.
// ---------------------------------------------------------------------------
__global__ void __launch_bounds__(256)
dwconv3x3(const float* __restrict__ in, const float* __restrict__ w9,
          float* __restrict__ out) {
  const int idx = blockIdx.x * 256 + threadIdx.x;   // B*C2*HW total
  const int pix = idx & (HW - 1);
  const int bc  = idx >> 14;
  const int c   = bc % C2;
  const int h = pix >> 7, w = pix & 127;
  const float* p  = in + (size_t)bc * HW;
  const float* wt = w9 + c * 9;
  float s = 0.f;
#pragma unroll
  for (int dy = -1; dy <= 1; ++dy)
#pragma unroll
    for (int dx = -1; dx <= 1; ++dx) {
      const int hh = h + dy, ww = w + dx;
      if ((unsigned)hh < (unsigned)IMH && (unsigned)ww < (unsigned)IMW)
        s += wt[(dy + 1) * 3 + (dx + 1)] * p[hh * IMW + ww];
    }
  out[idx] = s;
}

// ---------------------------------------------------------------------------
// Per-channel inverse L2 norms over hw for q (grid.y==0) and k (grid.y==1).
// ---------------------------------------------------------------------------
__global__ void __launch_bounds__(256)
rownorm(const float* __restrict__ q, const float* __restrict__ kv,
        float* __restrict__ invq, float* __restrict__ invk) {
  __shared__ float red[256];
  const int bc = blockIdx.x;            // b*192 + c
  const int which = blockIdx.y;
  const float* src;
  if (which == 0) src = q + (size_t)bc * HW;
  else {
    const int b = bc / DIMC, c = bc % DIMC;
    src = kv + ((size_t)b * C2 + c) * HW;
  }
  float s = 0.f;
  for (int i = threadIdx.x; i < HW; i += 256) { float v = src[i]; s += v * v; }
  red[threadIdx.x] = s;
  __syncthreads();
  for (int st = 128; st; st >>= 1) {
    if (threadIdx.x < st) red[threadIdx.x] += red[threadIdx.x + st];
    __syncthreads();
  }
  if (threadIdx.x == 0) {
    const float inv = 1.f / fmaxf(sqrtf(red[0]), 1e-12f);
    (which == 0 ? invq : invk)[bc] = inv;
  }
}

// ---------------------------------------------------------------------------
// attn = softmax( (q_raw . k_raw^T) * invq_c * invk_d * temp[h] ), 24x24
// padded to 32x32 bf16. One WG per (b,head); 8 waves split K = 16384.
// ---------------------------------------------------------------------------
__global__ void __launch_bounds__(256)
attn_wmma(const float* __restrict__ q, const float* __restrict__ kv,
          const float* __restrict__ invq, const float* __restrict__ invk,
          const float* __restrict__ temp, bf16* __restrict__ attn) {
  __shared__ float S[8][32][32];                     // 32 KB partials
  const int t = threadIdx.x, lane = t & 31, wv = t >> 5;
  const int b = blockIdx.x >> 3, h = blockIdx.x & 7;
  const float* qb = q  + ((size_t)b * DIMC + h * HD) * HW;
  const float* kb = kv + ((size_t)b * C2   + h * HD) * HW;

  const v8f zero8 = {0.f, 0.f, 0.f, 0.f, 0.f, 0.f, 0.f, 0.f};
  v8f acc[2][2];
#pragma unroll
  for (int f = 0; f < 2; ++f)
#pragma unroll
    for (int g = 0; g < 2; ++g) acc[f][g] = zero8;

  const int ml  = lane & 15;
  const int kbA = (lane < 16) ? 0 : 16;
  const int kbB = (lane < 16) ? 0 : 16;

  for (int it = 0; it < 64; ++it) {
    const int k0 = wv * 2048 + it * 32;
    v16bf a[2], bb[2];
#pragma unroll
    for (int f = 0; f < 2; ++f) {
      const int row = f * 16 + ml;
      if (row < HD) {
        const float* pr = qb + (size_t)row * HW + k0 + kbA;
        v8f lo = *(const v8f*)pr;
        v8f hi = *(const v8f*)(pr + 8);
#pragma unroll
        for (int i = 0; i < 8; ++i) { a[f][i] = (bf16)lo[i]; a[f][i + 8] = (bf16)hi[i]; }
      } else {
#pragma unroll
        for (int i = 0; i < 16; ++i) a[f][i] = (bf16)0.f;
      }
    }
#pragma unroll
    for (int g = 0; g < 2; ++g) {
      const int d = g * 16 + ml;
      if (d < HD) {
        const float* pr = kb + (size_t)d * HW + k0 + kbB;
        v8f lo = *(const v8f*)pr;
        v8f hi = *(const v8f*)(pr + 8);
#pragma unroll
        for (int i = 0; i < 8; ++i) { bb[g][i] = (bf16)lo[i]; bb[g][i + 8] = (bf16)hi[i]; }
      } else {
#pragma unroll
        for (int i = 0; i < 16; ++i) bb[g][i] = (bf16)0.f;
      }
    }
    acc[0][0] = wmma_bf16(a[0], bb[0], acc[0][0]);
    acc[0][1] = wmma_bf16(a[0], bb[1], acc[0][1]);
    acc[1][0] = wmma_bf16(a[1], bb[0], acc[1][0]);
    acc[1][1] = wmma_bf16(a[1], bb[1], acc[1][1]);
  }

  const int mh = (lane < 16) ? 0 : 8;
#pragma unroll
  for (int f = 0; f < 2; ++f)
#pragma unroll
    for (int g = 0; g < 2; ++g)
#pragma unroll
      for (int r = 0; r < 8; ++r)
        S[wv][f * 16 + mh + r][g * 16 + ml] = acc[f][g][r];
  __syncthreads();

  // cross-wave K reduction
  for (int cell = t; cell < 1024; cell += 256) {
    const int row = cell >> 5, col = cell & 31;
    float s = 0.f;
#pragma unroll
    for (int w = 0; w < 8; ++w) s += S[w][row][col];
    S[0][row][col] = s;     // each thread owns its cells: safe
  }
  __syncthreads();

  // scale + softmax over 24 valid cols, write 32x32 bf16 (zero padded)
  if (t < 32) {
    const int row = t;
    bf16* ab = attn + ((size_t)blockIdx.x * 32 + row) * 32;
    if (row < HD) {
      const float iq = invq[b * DIMC + h * HD + row];
      const float tm = temp[h];
      float vals[HD];
      float mx = -1e30f;
#pragma unroll
      for (int d = 0; d < HD; ++d) {
        vals[d] = S[0][row][d] * iq * invk[b * DIMC + h * HD + d] * tm;
        mx = fmaxf(mx, vals[d]);
      }
      float sum = 0.f;
#pragma unroll
      for (int d = 0; d < HD; ++d) { vals[d] = __expf(vals[d] - mx); sum += vals[d]; }
      const float inv = 1.f / sum;
#pragma unroll
      for (int d = 0; d < HD; ++d) ab[d] = (bf16)(vals[d] * inv);
#pragma unroll
      for (int d = HD; d < 32; ++d) ab[d] = (bf16)0.f;
    } else {
#pragma unroll
      for (int d = 0; d < 32; ++d) ab[d] = (bf16)0.f;
    }
  }
}

// ---------------------------------------------------------------------------
// out[c,e] = sum_d attn[c,d] * v[d,e] per (b,head). M=K=32 (24 valid), N=16384.
// grid.x = 64 (256-pixel tiles), grid.y = 64 (b*8+h). 8 waves x 32 cols.
// attn tile (2KB bf16, layout-exact) async-copied into LDS.
// ---------------------------------------------------------------------------
__global__ void __launch_bounds__(256)
av_wmma(const bf16* __restrict__ attn, const float* __restrict__ kv,
        float* __restrict__ out) {
  __shared__ __align__(16) bf16 Bt[256 * 32];        // v tile, transposed
  __shared__ __align__(16) bf16 Atile[32 * 32];      // attn tile
  const int t = threadIdx.x, lane = t & 31, wv = t >> 5;
  const int b = blockIdx.y >> 3, h = blockIdx.y & 7;
  const int p0 = blockIdx.x * 256;
  const float* vb = kv + ((size_t)b * C2 + DIMC + h * HD) * HW;
  const bf16* ab = attn + (size_t)blockIdx.y * 32 * 32;

  // attn tile: 1024 bf16 = 128 x b128; threads 0..127 copy 8 bf16 each
  if (t < 128) {
    const bf16* gp = ab + t * 8;
    bf16* lp = &Atile[t * 8];
#if defined(HAVE_ASYNC_LDS)
    ASYNC_CP_B128(gp, lp);
#else
    *(v8bf*)lp = *(const v8bf*)gp;
#endif
  }

  // stage v: Bt[n][k] = v[k, p0+n] (zero for padded k), packed b32 stores
#pragma unroll
  for (int kp = 0; kp < 16; ++kp) {
    const float v0 = (2 * kp     < HD) ? vb[(size_t)(2 * kp    ) * HW + p0 + t] : 0.f;
    const float v1 = (2 * kp + 1 < HD) ? vb[(size_t)(2 * kp + 1) * HW + p0 + t] : 0.f;
    *(unsigned*)&Bt[t * 32 + 2 * kp] = pack_bf16x2(v0, v1);
  }
#if defined(HAVE_ASYNC_LDS)
  ASYNC_WAIT();
#endif
  __syncthreads();

  v16bf a0 = lds_a_frag(Atile,            lane);
  v16bf a1 = lds_a_frag(Atile + 16 * 32,  lane);
  v16bf b0 = lds_b_frag(Bt + (wv * 32     ) * 32, lane);
  v16bf b1 = lds_b_frag(Bt + (wv * 32 + 16) * 32, lane);

  const v8f zero8 = {0.f, 0.f, 0.f, 0.f, 0.f, 0.f, 0.f, 0.f};
  v8f acc[2][2];
  acc[0][0] = wmma_bf16(a0, b0, zero8);
  acc[0][1] = wmma_bf16(a0, b1, zero8);
  acc[1][0] = wmma_bf16(a1, b0, zero8);
  acc[1][1] = wmma_bf16(a1, b1, zero8);

  const int ml = lane & 15;
  const int mh = (lane < 16) ? 0 : 8;
  float* ob = out + ((size_t)b * DIMC + h * HD) * HW;
#pragma unroll
  for (int f = 0; f < 2; ++f)
#pragma unroll
    for (int g = 0; g < 2; ++g)
#pragma unroll
      for (int r = 0; r < 8; ++r) {
        const int m = f * 16 + mh + r;
        if (m < HD)
          ob[(size_t)m * HW + p0 + wv * 32 + g * 16 + ml] = acc[f][g][r];
      }
}

// ---------------------------------------------------------------------------
extern "C" void kernel_launch(void* const* d_in, const int* in_sizes, int n_in,
                              void* d_out, int out_size, void* d_ws, size_t ws_size,
                              hipStream_t stream) {
  (void)in_sizes; (void)n_in; (void)out_size; (void)ws_size;
  const float* x      = (const float*)d_in[0];
  const float* y      = (const float*)d_in[1];
  const float* temp   = (const float*)d_in[2];
  const float* kv_w   = (const float*)d_in[3];
  const float* kv_dw  = (const float*)d_in[4];
  const float* q_w    = (const float*)d_in[5];
  const float* q_dw   = (const float*)d_in[6];
  const float* proj_w = (const float*)d_in[7];
  float* outp = (float*)d_out;

  // workspace layout (floats). kv_int aliases the dead q_int region; out_att
  // aliases the dead kv_int region. Peak ~= 5*SZ1*4B + ~1.2MB ~= 504 MB.
  const size_t SZ1 = (size_t)BATCH * DIMC * HW;      // 25,165,824 floats
  float* q_full  = (float*)d_ws;                     // [0, SZ1)
  float* q_int   = q_full + SZ1;                     // [SZ1, 2*SZ1)   dead after conv3x3
  float* kv_int  = q_int;                            // [SZ1, 3*SZ1)   dead after dw
  float* kv_full = q_full + 3 * SZ1;                 // [3*SZ1, 5*SZ1)
  float* out_att = kv_int;                           // reuse after dw

  float* smallf   = q_full + 5 * SZ1;
  bf16*  wkv_bf   = (bf16*)smallf;                   // 384*192
  bf16*  wq_bf    = wkv_bf  + C2 * DIMC;             // 192*192
  bf16*  wqdw_bf  = wq_bf   + DIMC * DIMC;           // 192*192*9
  bf16*  wproj_bf = wqdw_bf + DIMC * DIMC * 9;       // 192*192
  float* invq     = (float*)(wproj_bf + DIMC * DIMC);
  float* invk     = invq + BATCH * DIMC;
  bf16*  attn_bf  = (bf16*)(invk + BATCH * DIMC);    // 64*32*32

  // weight conversions
  cvt_f32_bf16<<<(C2 * DIMC + 255) / 256, 256, 0, stream>>>(kv_w, wkv_bf, C2 * DIMC);
  cvt_f32_bf16<<<(DIMC * DIMC + 255) / 256, 256, 0, stream>>>(q_w, wq_bf, DIMC * DIMC);
  cvt_f32_bf16<<<(DIMC * DIMC * 9 + 255) / 256, 256, 0, stream>>>(q_dw, wqdw_bf, DIMC * DIMC * 9);
  cvt_f32_bf16<<<(DIMC * DIMC + 255) / 256, 256, 0, stream>>>(proj_w, wproj_bf, DIMC * DIMC);

  // q chain first so q_int dies before kv_int aliases it
  conv1x1_wmma<<<dim3(BATCH * 128, 3), 256, 0, stream>>>(y, wq_bf, q_int, DIMC, DIMC);
  conv3x3_wmma<<<dim3(BATCH * IMH, 3), 256, 0, stream>>>(q_int, wqdw_bf, q_full);

  // kv chain
  conv1x1_wmma<<<dim3(BATCH * 128, 6), 256, 0, stream>>>(x, wkv_bf, kv_int, DIMC, C2);
  dwconv3x3<<<(BATCH * C2 * HW) / 256, 256, 0, stream>>>(kv_int, kv_dw, kv_full);

  // norms + attention
  rownorm<<<dim3(BATCH * DIMC, 2), 256, 0, stream>>>(q_full, kv_full, invq, invk);
  attn_wmma<<<BATCH * HEADS, 256, 0, stream>>>(q_full, kv_full, invq, invk, temp, attn_bf);
  av_wmma<<<dim3(HW / 256, BATCH * HEADS), 256, 0, stream>>>(attn_bf, kv_full, out_att);

  // final projection
  conv1x1_wmma<<<dim3(BATCH * 128, 3), 256, 0, stream>>>(out_att, wproj_bf, outp, DIMC, DIMC);
}
